// PartGLOM_64441689309573
// MI455X (gfx1250) — matrix operand order repaired and checked by reading, not compiled, not run on hardware
//
#include <hip/hip_runtime.h>
#include <map>
#include <string>
#include <vector>

typedef long long ll;
typedef __attribute__((ext_vector_type(16))) __bf16 v16bf;
typedef __attribute__((ext_vector_type(8)))  float  v8f;
typedef unsigned int u32x4 __attribute__((ext_vector_type(4)));

__device__ __forceinline__ unsigned short f2bf(float f) {
    unsigned int u = __float_as_uint(f);
    unsigned int r = u + 0x7FFFu + ((u >> 16) & 1u);   // RNE
    return (unsigned short)(r >> 16);
}

// ---------------------------------------------------------------------------
// Generic strided bf16-WMMA GEMM with fused bias / eval-BN / ReLU epilogue.
// Out[b,o,n] = act(BN( sum_c W[b,o,c] * X[b,c,n] + bias[o] ))
// Block: 256 threads = 8 waves (2 x 4); block tile 64(O) x 128(N); K step 32.
// Each wave computes a 32x32 output patch = 2x2 WMMA accumulators
// (4 v_wmma per K step fed by 8 ds_load_b128 fragment loads).
// Staging picks a coalesced thread->element map per operand via a uniform
// test of the fast stride (weights: K-contiguous; gram/activations:
// O/N-contiguous).  GUARD=true uses clamped addresses + multiply-by-mask so
// loads are never sunk into exec-predicated blocks (stays pipelined).
// Requires N % 128 == 0 (all call sites use N = 2048).
// ---------------------------------------------------------------------------
template <bool GUARD>
__global__ __launch_bounds__(256) void gemm_wmma_kernel(
    const float* __restrict__ Wm, ll wsO, ll wsC, ll wsB,
    const float* __restrict__ X,  ll xsC, ll xsN, ll xsB,
    float* __restrict__ Out, ll osO, ll osN, ll osB,
    const float* __restrict__ bias,
    const float* __restrict__ gg, const float* __restrict__ gbeta,
    const float* __restrict__ gmean, const float* __restrict__ gvar,
    int relu, int O, int C, int N)
{
    __shared__ __align__(16) unsigned short sA[64][32];   // [o_local][k]  4 KB
    __shared__ __align__(16) unsigned short sB[128][32];  // [n_local][k]  8 KB

    const int tid   = threadIdx.x;
    const int b     = blockIdx.z;
    const int oBase = blockIdx.y * 64;
    const int nBase = blockIdx.x * 128;
    const int wid   = tid >> 5, lane = tid & 31;
    const int half  = lane >> 4, lrow = lane & 15;
    const int oW = (wid >> 2) * 32;   // wave origin in block tile (O)
    const int nW = (wid & 3) * 32;    // wave origin in block tile (N)

    const float* Wb = Wm + (ll)b * wsB;
    const float* Xb = X  + (ll)b * xsB;

    const bool aKfast = (wsC == 1);   // uniform: weights K-contiguous
    const bool bNfast = (xsN == 1);   // uniform: activations N-contiguous

    v8f acc[2][2];
    const v8f vzero = {0.f, 0.f, 0.f, 0.f, 0.f, 0.f, 0.f, 0.f};
    acc[0][0] = vzero; acc[0][1] = vzero; acc[1][0] = vzero; acc[1][1] = vzero;

    for (int k0 = 0; k0 < C; k0 += 32) {
        // ---- phase 1: gather tiles into registers (loads pipeline) ----
        float va[8];
        if (aKfast) {                      // lanes walk K (stride wsC==1)
            #pragma unroll
            for (int t = 0; t < 8; ++t) {
                int idx = tid + t * 256;
                int go = oBase + (idx >> 5), gc = k0 + (idx & 31);
                if (GUARD) {
                    float v = Wb[(ll)min(go, O - 1) * wsO + (ll)min(gc, C - 1)];
                    va[t] = v * ((go < O && gc < C) ? 1.f : 0.f);
                } else {
                    va[t] = Wb[(ll)go * wsO + (ll)gc];
                }
            }
        } else {                           // lanes walk O (gram: wsO==1)
            #pragma unroll
            for (int t = 0; t < 8; ++t) {
                int idx = tid + t * 256;
                int go = oBase + (idx & 63), gc = k0 + (idx >> 6);
                if (GUARD) {
                    float v = Wb[(ll)min(go, O - 1) * wsO + (ll)min(gc, C - 1) * wsC];
                    va[t] = v * ((go < O && gc < C) ? 1.f : 0.f);
                } else {
                    va[t] = Wb[(ll)go * wsO + (ll)gc * wsC];
                }
            }
        }
        float vb[16];
        if (bNfast) {                      // lanes walk N (coalesced 4B)
            #pragma unroll
            for (int t = 0; t < 16; ++t) {
                int idx = tid + t * 256;
                int gn = nBase + (idx & 127), gc = k0 + (idx >> 7);
                if (GUARD) {
                    float v = Xb[(ll)min(gc, C - 1) * xsC + (ll)gn];
                    vb[t] = v * ((gc < C) ? 1.f : 0.f);
                } else {
                    vb[t] = Xb[(ll)gc * xsC + (ll)gn];
                }
            }
        } else {
            #pragma unroll
            for (int t = 0; t < 16; ++t) {
                int idx = tid + t * 256;
                int gn = nBase + (idx >> 5), gc = k0 + (idx & 31);
                if (GUARD) {
                    float v = Xb[(ll)min(gc, C - 1) * xsC + (ll)gn * xsN];
                    vb[t] = v * ((gc < C) ? 1.f : 0.f);
                } else {
                    vb[t] = Xb[(ll)gc * xsC + (ll)gn * xsN];
                }
            }
        }
        if (!GUARD && (k0 + 64 <= C)) {    // uniform-branch prefetch of next K chunk
            int gn = nBase + (tid & 127), gc = k0 + 32 + (tid >> 7);
            __builtin_prefetch(&Xb[(ll)gc * xsC + (ll)gn * xsN], 0, 1);
        }
        // ---- phase 2: convert + store to LDS (same maps as phase 1) ----
        if (aKfast) {
            #pragma unroll
            for (int t = 0; t < 8; ++t) {
                int idx = tid + t * 256;
                sA[idx >> 5][idx & 31] = f2bf(va[t]);
            }
        } else {
            #pragma unroll
            for (int t = 0; t < 8; ++t) {
                int idx = tid + t * 256;
                sA[idx & 63][idx >> 6] = f2bf(va[t]);
            }
        }
        if (bNfast) {
            #pragma unroll
            for (int t = 0; t < 16; ++t) {
                int idx = tid + t * 256;
                sB[idx & 127][idx >> 7] = f2bf(vb[t]);
            }
        } else {
            #pragma unroll
            for (int t = 0; t < 16; ++t) {
                int idx = tid + t * 256;
                sB[idx >> 5][idx & 31] = f2bf(vb[t]);
            }
        }
        __syncthreads();

        // ---- phase 3: fragments + 4 WMMAs ----
        union V16 { v16bf v; u32x4 q[2]; } af[2], bf[2];
        #pragma unroll
        for (int i = 0; i < 2; ++i) {
            // A: lane row M=lrow; elems 0..7 -> K=8*half+0..7 ; 8..15 -> +16
            af[i].q[0] = *(const u32x4*)&sA[oW + i * 16 + lrow][8 * half];
            af[i].q[1] = *(const u32x4*)&sA[oW + i * 16 + lrow][16 + 8 * half];
            // B: lane col N=lrow; elems 0..7 -> K=8*half+0..7 ; 8..15 -> +16
            bf[i].q[0] = *(const u32x4*)&sB[nW + i * 16 + lrow][8 * half];
            bf[i].q[1] = *(const u32x4*)&sB[nW + i * 16 + lrow][16 + 8 * half];
        }
        #pragma unroll
        for (int i = 0; i < 2; ++i)
            #pragma unroll
            for (int j = 0; j < 2; ++j)
                acc[i][j] = __builtin_amdgcn_wmma_f32_16x16x32_bf16(
                                false, af[i].v, false, bf[j].v, (short)0,
                                acc[i][j], false, false);
        __syncthreads();
    }

    // ---- epilogue: D layout vgpr r -> M = r + 8*half ; N = lrow ----
    #pragma unroll
    for (int i = 0; i < 2; ++i) {
        #pragma unroll
        for (int j = 0; j < 2; ++j) {
            int n = nBase + nW + j * 16 + lrow;
            #pragma unroll
            for (int r = 0; r < 8; ++r) {
                int o = oBase + oW + i * 16 + r + 8 * half;
                float v = acc[i][j][r];
                if (bias) v += bias[GUARD ? min(o, O - 1) : o];
                if (gg) {
                    int oc = GUARD ? min(o, O - 1) : o;
                    v = (v - gmean[oc]) * (gg[oc] * rsqrtf(gvar[oc] + 1e-5f)) + gbeta[oc];
                }
                if (relu) v = fmaxf(v, 0.f);
                if (GUARD) {
                    if (o < O) Out[(ll)b * osB + (ll)o * osO + (ll)n * osN] = v;
                } else {
                    Out[(ll)b * osB + (ll)o * osO + (ll)n * osN] = v;
                }
            }
        }
    }
}

// ----- small dense: out[b,o] = act(BN(in[b,:]·W[o,:] + bias)) (+eye) --------
__global__ void dense_kernel(const float* __restrict__ In, int I,
                             const float* __restrict__ W, const float* __restrict__ bias,
                             const float* __restrict__ gg, const float* __restrict__ gbeta,
                             const float* __restrict__ gmean, const float* __restrict__ gvar,
                             int relu, int eyeC, float* __restrict__ Out, int O, int Bc)
{
    int t = blockIdx.x * blockDim.x + threadIdx.x;
    if (t >= O * Bc) return;
    int b = t / O, o = t % O;
    const float* in = In + (ll)b * I;
    const float* w  = W + (ll)o * I;
    float v = 0.f;
    for (int i = 0; i < I; ++i) v += in[i] * w[i];
    if (bias) v += bias[o];
    if (gg)   v = (v - gmean[o]) * (gg[o] * rsqrtf(gvar[o] + 1e-5f)) + gbeta[o];
    if (relu) v = fmaxf(v, 0.f);
    if (eyeC > 0 && (o / eyeC) == (o % eyeC)) v += 1.f;
    Out[t] = v;
}

// ----- global max pool over points ------------------------------------------
__global__ void maxpool_kernel(const float* __restrict__ In, float* __restrict__ Out, int N)
{
    int c = blockIdx.x, b = blockIdx.y, C = gridDim.x;
    const float* p = In + ((ll)b * C + c) * N;
    float m = -3.4e38f;
    for (int i = threadIdx.x; i < N; i += blockDim.x) m = fmaxf(m, p[i]);
    __shared__ float red[256];
    red[threadIdx.x] = m; __syncthreads();
    for (int s = 128; s > 0; s >>= 1) {
        if (threadIdx.x < s) red[threadIdx.x] = fmaxf(red[threadIdx.x], red[threadIdx.x + s]);
        __syncthreads();
    }
    if (threadIdx.x == 0) Out[(ll)b * C + c] = red[0];
}

// ----- xc[b,d,n] = sum_c x[b,n,c] * t3[b,c,d] -------------------------------
__global__ void apply_t3_kernel(const float* __restrict__ x, const float* __restrict__ t3,
                                float* __restrict__ xc, int N)
{
    int t = blockIdx.x * blockDim.x + threadIdx.x;
    int b = t / N, n = t % N;
    const float* xp = x + ((ll)b * N + n) * 3;
    const float* tp = t3 + (ll)b * 9;
    float* op = xc + (ll)b * 3 * N + n;
    #pragma unroll
    for (int d = 0; d < 3; ++d)
        op[(ll)d * N] = xp[0] * tp[d] + xp[1] * tp[3 + d] + xp[2] * tp[6 + d];
}

// ----- kNN (K=20) on xyz: per-thread scan with stable insertion -------------
__global__ void knn_xyz_kernel(const float* __restrict__ x, int* __restrict__ idx, int N, int Bc)
{
    int t = blockIdx.x * blockDim.x + threadIdx.x;
    if (t >= N * Bc) return;
    int b = t / N, n = t % N;
    const float* xb = x + (ll)b * N * 3;
    float px = xb[(ll)n * 3], py = xb[(ll)n * 3 + 1], pz = xb[(ll)n * 3 + 2];
    float bd[20]; int bi[20];
    #pragma unroll
    for (int j = 0; j < 20; ++j) { bd[j] = 3.4e38f; bi[j] = 0; }
    for (int m = 0; m < N; ++m) {
        if (m == n) continue;  // self excluded (+1e10 in reference)
        float dx = px - xb[(ll)m * 3], dy = py - xb[(ll)m * 3 + 1], dz = pz - xb[(ll)m * 3 + 2];
        float d = dx * dx + dy * dy + dz * dz;
        if (d < bd[19]) {
            int p = 19;
            while (p > 0 && bd[p - 1] > d) { bd[p] = bd[p - 1]; bi[p] = bi[p - 1]; --p; }
            bd[p] = d; bi[p] = m;
        }
    }
    for (int j = 0; j < 20; ++j) idx[(ll)t * 20 + j] = bi[j] + b * N;
}

// ----- sq[p] = sum_c X[b,c,n]^2 (X strided) ---------------------------------
__global__ void sqnorm_kernel(const float* __restrict__ X, ll bStride, int C, int N,
                              float* __restrict__ sq, int BN)
{
    int t = blockIdx.x * blockDim.x + threadIdx.x;
    if (t >= BN) return;
    int b = t / N, n = t % N;
    const float* p = X + (ll)b * bStride + n;
    float s = 0.f;
    for (int c = 0; c < C; ++c) { float v = p[(ll)c * N]; s += v * v; }
    sq[t] = s;
}

// ----- kNN (K=20) on embeddings from precomputed Gram row -------------------
__global__ void knn_emb_kernel(const float* __restrict__ G, const float* __restrict__ sq,
                               int* __restrict__ idx, int N, int b)
{
    int n = blockIdx.x * blockDim.x + threadIdx.x;
    if (n >= N) return;
    float sn = sq[n];
    const float* g = G + (ll)n * N;
    float bd[20]; int bi[20];
    #pragma unroll
    for (int j = 0; j < 20; ++j) { bd[j] = 3.4e38f; bi[j] = 0; }
    for (int m = 0; m < N; ++m) {
        if (m == n) continue;
        float d = sn + sq[m] - 2.f * g[m];
        if (d < bd[19]) {
            int p = 19;
            while (p > 0 && bd[p - 1] > d) { bd[p] = bd[p - 1]; bi[p] = bi[p - 1]; --p; }
            bd[p] = d; bi[p] = m;
        }
    }
    for (int j = 0; j < 20; ++j) idx[((ll)(b * N + n)) * 20 + j] = bi[j] + b * N;
}

// ----- center-major edge build: dedupe (coalesce) + normal-cosine weight ----
__global__ void build_edges_kernel(const int* __restrict__ idx1, const int* __restrict__ idx2,
                                   const float* __restrict__ normals,
                                   int* __restrict__ er, int* __restrict__ ev,
                                   float* __restrict__ ew, int BN)
{
    int c = blockIdx.x * blockDim.x + threadIdx.x;
    if (c >= BN) return;
    int r[40];
    for (int j = 0; j < 20; ++j) r[j]      = idx1[(ll)c * 20 + j];
    for (int j = 0; j < 20; ++j) r[20 + j] = idx2[(ll)c * 20 + j];
    float ncx = normals[(ll)c * 3], ncy = normals[(ll)c * 3 + 1], ncz = normals[(ll)c * 3 + 2];
    float ncn = sqrtf(ncx * ncx + ncy * ncy + ncz * ncz);
    for (int j = 0; j < 40; ++j) {
        int v = 1;
        for (int j2 = 0; j2 < j; ++j2) if (r[j2] == r[j]) { v = 0; break; }
        int ri = r[j];
        float nx = normals[(ll)ri * 3], ny = normals[(ll)ri * 3 + 1], nz = normals[(ll)ri * 3 + 2];
        float nrn = sqrtf(nx * nx + ny * ny + nz * nz);
        float cosv = fabsf(nx * ncx + ny * ncy + nz * ncz) / fmaxf(nrn * ncn, 1e-8f);
        ll e = (ll)c * 40 + j;
        er[e] = ri; ev[e] = v; ew[e] = cosv * (float)v;
    }
}

// ----- smoothing stage 1: per-point label + lg @ Wg -------------------------
__global__ void smooth_pre_kernel(const float* __restrict__ mlog, const float* __restrict__ gw,
                                  float* __restrict__ xw, int* __restrict__ lab, int N, int BN)
{
    int p = blockIdx.x * blockDim.x + threadIdx.x;
    if (p >= BN) return;
    int b = p / N, n = p % N;
    const float* base = mlog + (ll)b * 16 * N + n;
    float lg[16];
    #pragma unroll
    for (int l = 0; l < 16; ++l) lg[l] = base[(ll)l * N];
    int best = 0; float bv = lg[0];
    #pragma unroll
    for (int l = 1; l < 16; ++l) if (lg[l] > bv) { bv = lg[l]; best = l; }
    lab[p] = best;
    #pragma unroll
    for (int l = 0; l < 16; ++l) {
        float s = 0.f;
        #pragma unroll
        for (int j = 0; j < 16; ++j) s += lg[j] * gw[j * 16 + l];
        xw[(ll)p * 16 + l] = s;
    }
}

// ----- smoothing stage 2: votes -> freq -> kept weights -> degree -----------
__global__ void smooth_deg_kernel(const int* __restrict__ er, const int* __restrict__ ev,
                                  const float* __restrict__ ew, const int* __restrict__ lab,
                                  float* __restrict__ kw, float* __restrict__ dis, int BN)
{
    int c = blockIdx.x * blockDim.x + threadIdx.x;
    if (c >= BN) return;
    float votes[16];
    #pragma unroll
    for (int l = 0; l < 16; ++l) votes[l] = 0.f;
    for (int j = 0; j < 40; ++j) {
        ll e = (ll)c * 40 + j;
        if (ev[e]) votes[lab[er[e]]] += 1.f;
    }
    int freq = 0; float fv = votes[0];
    #pragma unroll
    for (int l = 1; l < 16; ++l) if (votes[l] > fv) { fv = votes[l]; freq = l; }
    float deg = 1.f;  // self loop
    for (int j = 0; j < 40; ++j) {
        ll e = (ll)c * 40 + j;
        float w = (ev[e] && (lab[er[e]] == freq)) ? ew[e] : 0.f;
        kw[e] = w; deg += w;
    }
    dis[c] = rsqrtf(deg);
}

// ----- smoothing stage 3: GCN aggregate -> lm + argmax sample ---------------
__global__ void smooth_gcn_kernel(const int* __restrict__ er, const float* __restrict__ kw,
                                  const float* __restrict__ dis, const float* __restrict__ xw,
                                  const float* __restrict__ gb, float* __restrict__ lm,
                                  int* __restrict__ sel, int BN)
{
    int c = blockIdx.x * blockDim.x + threadIdx.x;
    if (c >= BN) return;
    float dc = dis[c];
    float acc[16];
    #pragma unroll
    for (int l = 0; l < 16; ++l) acc[l] = dc * dc * xw[(ll)c * 16 + l] + gb[l];
    for (int j = 0; j < 40; ++j) {
        ll e = (ll)c * 40 + j;
        int ri = er[e];
        float nw = dis[ri] * kw[e] * dc;
        #pragma unroll
        for (int l = 0; l < 16; ++l) acc[l] += nw * xw[(ll)ri * 16 + l];
    }
    int best = 0; float bv = acc[0];
    #pragma unroll
    for (int l = 1; l < 16; ++l) if (acc[l] > bv) { bv = acc[l]; best = l; }
    #pragma unroll
    for (int l = 0; l < 16; ++l) lm[(ll)c * 16 + l] = acc[l];
    sel[c] = best;
}

// ----- latent: emb2[b,k,n] = feat[p]·latw[sel[p],:,k] + latb[sel[p],k] ------
__global__ void latent_kernel(const float* __restrict__ feat, ll fB,
                              const float* __restrict__ lw, const float* __restrict__ lb,
                              const int* __restrict__ sel, float* __restrict__ dst, ll dB,
                              int N, int BN)
{
    int t = blockIdx.x * blockDim.x + threadIdx.x;
    int k = t & 63; int p = t >> 6;
    if (p >= BN) return;
    int b = p / N, n = p % N;
    int l = sel[p];
    const float* w = lw + (ll)l * 512 * 64 + k;
    const float* f = feat + (ll)b * fB + n;
    float s = lb[(ll)l * 64 + k];
    for (int ci = 0; ci < 512; ++ci) s += f[(ll)ci * N] * w[(ll)ci * 64];
    dst[(ll)b * dB + (ll)k * N + n] = s;
}

// ----- strided (B,C,N) copy --------------------------------------------------
__global__ void copy_ch_kernel(const float* __restrict__ src, ll sB,
                               float* __restrict__ dst, ll dB, int C, int N, ll total)
{
    ll t = (ll)blockIdx.x * blockDim.x + threadIdx.x;
    if (t >= total) return;
    int n = (int)(t % N); ll r = t / N; int c = (int)(r % C); int b = (int)(r / C);
    dst[(ll)b * dB + (ll)c * N + n] = src[(ll)b * sB + (ll)c * N + n];
}

// ===========================================================================
// host side
// ===========================================================================
static void launch_gemm(hipStream_t s, int Bc, int O, int C, int N,
                        const float* W, ll wsO, ll wsC, ll wsB,
                        const float* X, ll xsC, ll xsN, ll xsB,
                        float* Out, ll osO, ll osN, ll osB,
                        const float* bias, const float* gg, const float* gbeta,
                        const float* gmean, const float* gvar, int relu)
{
    dim3 grid(N / 128, (O + 63) / 64, Bc);
    bool guard = (O % 64 != 0) || (C % 32 != 0);
    if (guard)
        gemm_wmma_kernel<true><<<grid, 256, 0, s>>>(W, wsO, wsC, wsB, X, xsC, xsN, xsB,
                                                    Out, osO, osN, osB, bias,
                                                    gg, gbeta, gmean, gvar, relu, O, C, N);
    else
        gemm_wmma_kernel<false><<<grid, 256, 0, s>>>(W, wsO, wsC, wsB, X, xsC, xsN, xsB,
                                                     Out, osO, osN, osB, bias,
                                                     gg, gbeta, gmean, gvar, relu, O, C, N);
}

extern "C" void kernel_launch(void* const* d_in, const int* in_sizes, int n_in,
                              void* d_out, int out_size, void* d_ws, size_t ws_size,
                              hipStream_t stream) {
    const int Nn = 2048, Bc = 8, BN = Bc * Nn;
    if (n_in < 134) return;

    // ---- detect pytree flattening order (insertion vs sorted keys) ----
    bool topIns = (in_sizes[1] == Bc * Nn * 3);     // normals at [1] => top-level insertion
    int  pStart = topIns ? 2 : 1;
    bool pSorted = (in_sizes[pStart] == 64);        // sorted params start with bn1.b (64)

    std::vector<std::string> names;
    auto cp = [&](const std::string& nm) {
        if (pSorted) { names.push_back(nm + ".b"); names.push_back(nm + ".w"); }
        else         { names.push_back(nm + ".w"); names.push_back(nm + ".b"); }
    };
    auto bp = [&](const std::string& nm) {
        if (pSorted) { names.push_back(nm + ".b"); names.push_back(nm + ".g");
                       names.push_back(nm + ".m"); names.push_back(nm + ".v"); }
        else         { names.push_back(nm + ".g"); names.push_back(nm + ".b");
                       names.push_back(nm + ".m"); names.push_back(nm + ".v"); }
    };
    auto tnet = [&](const std::string& s) {
        if (pSorted) { bp(s+".b1"); bp(s+".b2"); bp(s+".b3"); bp(s+".bf1"); bp(s+".bf2");
                       cp(s+".c1"); cp(s+".c2"); cp(s+".c3"); cp(s+".f1"); cp(s+".f2"); cp(s+".f3"); }
        else         { cp(s+".c1"); cp(s+".c2"); cp(s+".c3"); bp(s+".b1"); bp(s+".b2"); bp(s+".b3");
                       cp(s+".f1"); cp(s+".f2"); cp(s+".f3"); bp(s+".bf1"); bp(s+".bf2"); }
    };
    if (pSorted) {
        bp("bn1"); bp("bn2"); bp("bn3"); bp("bn4"); bp("bn5");
        cp("conv1"); cp("conv2"); cp("conv3"); cp("conv4"); cp("conv5");
        cp("gcn"); cp("latent");
        bp("midcls.b1"); cp("midcls.c1"); cp("midcls.c2");
        bp("tb1"); bp("tb2"); bp("tb3");
        cp("tc1"); cp("tc2"); cp("tc3");
        tnet("tnet3"); tnet("tnetk");
        bp("topcls.b1"); cp("topcls.c1"); cp("topcls.c2");
    } else {
        tnet("tnet3"); tnet("tnetk");
        cp("conv1"); cp("conv2"); cp("conv3"); cp("conv4"); cp("conv5");
        bp("bn1"); bp("bn2"); bp("bn3"); bp("bn4"); bp("bn5");
        cp("midcls.c1"); bp("midcls.b1"); cp("midcls.c2");
        cp("gcn"); cp("latent");
        cp("tc1"); cp("tc2"); cp("tc3"); bp("tb1"); bp("tb2"); bp("tb3");
        cp("topcls.c1"); bp("topcls.b1"); cp("topcls.c2");
    }
    std::map<std::string, const float*> pm;
    for (size_t i = 0; i < names.size(); ++i) pm[names[i]] = (const float*)d_in[pStart + (int)i];
    auto g = [&](const char* nm) -> const float* { auto it = pm.find(nm); return it == pm.end() ? nullptr : it->second; };

    const float* x       = (const float*)d_in[topIns ? 0 : (n_in - 1)];
    const float* normals = (const float*)d_in[topIns ? 1 : 0];

    // ---- workspace arena (float elements) ----
    const ll OFF_BIG = 0;                                   // 1216*N*B (tnet hidden / hcat)
    const ll OFF_P   = OFF_BIG + (ll)1216 * Nn * Bc;        // 2048*2048 gram / tc1 out
    const ll OFF_Q   = OFF_P + (ll)2048 * 2048;             // mid c1 / tc2 out
    const ll OFF_XC  = OFF_Q + (ll)256 * Nn * Bc;
    const ll OFF_XK  = OFF_XC + (ll)3 * Nn * Bc;
    const ll OFF_G   = OFF_XK + (ll)128 * Nn * Bc;
    const ll OFF_F1  = OFF_G + (ll)1024 * Bc;
    const ll OFF_F2  = OFF_F1 + (ll)512 * Bc;
    const ll OFF_T3  = OFF_F2 + (ll)256 * Bc;
    const ll OFF_TK  = OFF_T3 + 128;
    const ll OFF_SQ  = OFF_TK + (ll)128 * 128 * Bc;
    const ll OFF_I1  = OFF_SQ + BN;
    const ll OFF_I2  = OFF_I1 + (ll)BN * 20;
    const ll OFF_ER  = OFF_I2 + (ll)BN * 20;
    const ll OFF_EV  = OFF_ER + (ll)BN * 40;
    const ll OFF_EW  = OFF_EV + (ll)BN * 40;
    const ll OFF_KW  = OFF_EW + (ll)BN * 40;
    const ll OFF_ML  = OFF_KW + (ll)BN * 40;
    const ll OFF_XW  = OFF_ML + (ll)16 * Nn * Bc;
    const ll OFF_LAB = OFF_XW + (ll)BN * 16;
    const ll OFF_DIS = OFF_LAB + BN;
    const ll OFF_SEL = OFF_DIS + BN;
    const ll OFF_TP1 = OFF_SEL + BN;
    const ll WS_TOT  = OFF_TP1 + (ll)64 * Nn * Bc;
    if (ws_size < (size_t)WS_TOT * sizeof(float)) return;

    float* ws  = (float*)d_ws;
    float* out = (float*)d_out;
    int* i1  = (int*)(ws + OFF_I1);
    int* i2  = (int*)(ws + OFF_I2);
    int* er  = (int*)(ws + OFF_ER);
    int* ev  = (int*)(ws + OFF_EV);
    int* lab = (int*)(ws + OFF_LAB);
    int* sel = (int*)(ws + OFF_SEL);

    // d_out layout (floats)
    const ll OUT_H     = (ll)Bc * 1088 * Nn;          // 17,825,792
    const ll OUT_LT    = OUT_H + (ll)Bc * 128 * Nn;
    const ll OUT_LM    = OUT_LT + (ll)BN * 50;
    const ll OUT_OUT5  = OUT_LM + (ll)BN * 16;

    float* H1 = ws + OFF_BIG;
    float* H2 = H1 + (ll)Bc * 64 * Nn;
    float* H3 = H2 + (ll)Bc * 128 * Nn;
    const ll EB = (ll)1088 * Nn;                      // embed per-batch stride

    #define BNPTRS(nm) g(nm ".g"), g(nm ".b"), g(nm ".m"), g(nm ".v")
    #define NOBN nullptr, nullptr, nullptr, nullptr

    // 1) kNN on xyz
    knn_xyz_kernel<<<BN / 256, 256, 0, stream>>>(x, i1, Nn, Bc);

    // 2) tnet3 (input = x^T via strides: (b,c,n) -> x[b*N*3 + n*3 + c])
    launch_gemm(stream, Bc, 64, 3, Nn, g("tnet3.c1.w"), 3, 1, 0,
                x, 1, 3, (ll)Nn * 3, H1, Nn, 1, (ll)64 * Nn,
                g("tnet3.c1.b"), BNPTRS("tnet3.b1"), 1);
    launch_gemm(stream, Bc, 128, 64, Nn, g("tnet3.c2.w"), 64, 1, 0,
                H1, Nn, 1, (ll)64 * Nn, H2, Nn, 1, (ll)128 * Nn,
                g("tnet3.c2.b"), BNPTRS("tnet3.b2"), 1);
    launch_gemm(stream, Bc, 1024, 128, Nn, g("tnet3.c3.w"), 128, 1, 0,
                H2, Nn, 1, (ll)128 * Nn, H3, Nn, 1, (ll)1024 * Nn,
                g("tnet3.c3.b"), BNPTRS("tnet3.b3"), 1);
    maxpool_kernel<<<dim3(1024, Bc), 256, 0, stream>>>(H3, ws + OFF_G, Nn);
    dense_kernel<<<(Bc * 512 + 255) / 256, 256, 0, stream>>>(ws + OFF_G, 1024,
        g("tnet3.f1.w"), g("tnet3.f1.b"), BNPTRS("tnet3.bf1"), 1, 0, ws + OFF_F1, 512, Bc);
    dense_kernel<<<(Bc * 256 + 255) / 256, 256, 0, stream>>>(ws + OFF_F1, 512,
        g("tnet3.f2.w"), g("tnet3.f2.b"), BNPTRS("tnet3.bf2"), 1, 0, ws + OFF_F2, 256, Bc);
    dense_kernel<<<(Bc * 9 + 255) / 256, 256, 0, stream>>>(ws + OFF_F2, 256,
        g("tnet3.f3.w"), g("tnet3.f3.b"), NOBN, 0, 3, ws + OFF_T3, 9, Bc);

    // 3) xc = x @ t3 (transposed layout)
    apply_t3_kernel<<<BN / 256, 256, 0, stream>>>(x, ws + OFF_T3, ws + OFF_XC, Nn);

    // 4) conv1..conv3 -> embed channels [0,64), [64,192), [192,320)
    launch_gemm(stream, Bc, 64, 3, Nn, g("conv1.w"), 3, 1, 0,
                ws + OFF_XC, Nn, 1, (ll)3 * Nn, out, Nn, 1, EB,
                g("conv1.b"), BNPTRS("bn1"), 1);
    launch_gemm(stream, Bc, 128, 64, Nn, g("conv2.w"), 64, 1, 0,
                out, Nn, 1, EB, out + (ll)64 * Nn, Nn, 1, EB,
                g("conv2.b"), BNPTRS("bn2"), 1);
    launch_gemm(stream, Bc, 128, 128, Nn, g("conv3.w"), 128, 1, 0,
                out + (ll)64 * Nn, Nn, 1, EB, out + (ll)192 * Nn, Nn, 1, EB,
                g("conv3.b"), BNPTRS("bn3"), 1);

    // 5) tnetk on out3
    launch_gemm(stream, Bc, 64, 128, Nn, g("tnetk.c1.w"), 128, 1, 0,
                out + (ll)192 * Nn, Nn, 1, EB, H1, Nn, 1, (ll)64 * Nn,
                g("tnetk.c1.b"), BNPTRS("tnetk.b1"), 1);
    launch_gemm(stream, Bc, 128, 64, Nn, g("tnetk.c2.w"), 64, 1, 0,
                H1, Nn, 1, (ll)64 * Nn, H2, Nn, 1, (ll)128 * Nn,
                g("tnetk.c2.b"), BNPTRS("tnetk.b2"), 1);
    launch_gemm(stream, Bc, 1024, 128, Nn, g("tnetk.c3.w"), 128, 1, 0,
                H2, Nn, 1, (ll)128 * Nn, H3, Nn, 1, (ll)1024 * Nn,
                g("tnetk.c3.b"), BNPTRS("tnetk.b3"), 1);
    maxpool_kernel<<<dim3(1024, Bc), 256, 0, stream>>>(H3, ws + OFF_G, Nn);
    dense_kernel<<<(Bc * 512 + 255) / 256, 256, 0, stream>>>(ws + OFF_G, 1024,
        g("tnetk.f1.w"), g("tnetk.f1.b"), BNPTRS("tnetk.bf1"), 1, 0, ws + OFF_F1, 512, Bc);
    dense_kernel<<<(Bc * 256 + 255) / 256, 256, 0, stream>>>(ws + OFF_F1, 512,
        g("tnetk.f2.w"), g("tnetk.f2.b"), BNPTRS("tnetk.bf2"), 1, 0, ws + OFF_F2, 256, Bc);
    dense_kernel<<<(Bc * 16384 + 255) / 256, 256, 0, stream>>>(ws + OFF_F2, 256,
        g("tnetk.f3.w"), g("tnetk.f3.b"), NOBN, 0, 128, ws + OFF_TK, 16384, Bc);

    // 6) xk[b,d,n] = sum_c out3[b,c,n]*tk[b,c,d]  (W = tk transposed via strides)
    launch_gemm(stream, Bc, 128, 128, Nn, ws + OFF_TK, 1, 128, 16384,
                out + (ll)192 * Nn, Nn, 1, EB, ws + OFF_XK, Nn, 1, (ll)128 * Nn,
                nullptr, NOBN, 0);

    // 7) conv4, conv5 (conv5: BN, no ReLU) -> embed [320,576), [576,1088)
    launch_gemm(stream, Bc, 256, 128, Nn, g("conv4.w"), 128, 1, 0,
                ws + OFF_XK, Nn, 1, (ll)128 * Nn, out + (ll)320 * Nn, Nn, 1, EB,
                g("conv4.b"), BNPTRS("bn4"), 1);
    launch_gemm(stream, Bc, 512, 256, Nn, g("conv5.w"), 256, 1, 0,
                out + (ll)320 * Nn, Nn, 1, EB, out + (ll)576 * Nn, Nn, 1, EB,
                g("conv5.b"), BNPTRS("bn5"), 0);

    // 8) copy out5 to its own output slot
    {
        ll total = (ll)Bc * 512 * Nn;
        copy_ch_kernel<<<(unsigned)((total + 255) / 256), 256, 0, stream>>>(
            out + (ll)576 * Nn, EB, out + OUT_OUT5, (ll)512 * Nn, 512, Nn, total);
    }

    // 9) squared norms of embedding columns
    sqnorm_kernel<<<BN / 256, 256, 0, stream>>>(out + (ll)576 * Nn, EB, 512, Nn, ws + OFF_SQ, BN);

    // 10) per-batch Gram (WMMA, unguarded fast path, both operands coalesced)
    for (int b = 0; b < Bc; ++b) {
        const float* o5b = out + (ll)b * EB + (ll)576 * Nn;
        launch_gemm(stream, 1, Nn, 512, Nn, o5b, 1, Nn, 0,
                    o5b, Nn, 1, 0, ws + OFF_P, Nn, 1, 0, nullptr, NOBN, 0);
        knn_emb_kernel<<<Nn / 256, 256, 0, stream>>>(ws + OFF_P, ws + OFF_SQ + (ll)b * Nn, i2, Nn, b);
    }

    // 11) edges: dedupe + cosine weights (fixed degree 40 per center)
    build_edges_kernel<<<BN / 256, 256, 0, stream>>>(i1, i2, normals, er, ev, ws + OFF_EW, BN);

    // 12) mid classifier
    launch_gemm(stream, Bc, 256, 512, Nn, g("midcls.c1.w"), 512, 1, 0,
                out + (ll)576 * Nn, Nn, 1, EB, ws + OFF_Q, Nn, 1, (ll)256 * Nn,
                g("midcls.c1.b"), BNPTRS("midcls.b1"), 1);
    launch_gemm(stream, Bc, 16, 256, Nn, g("midcls.c2.w"), 256, 1, 0,
                ws + OFF_Q, Nn, 1, (ll)256 * Nn, ws + OFF_ML, Nn, 1, (ll)16 * Nn,
                g("midcls.c2.b"), NOBN, 0);

    // 13) majority-vote smoothing + GCN -> lm (+ hard sample)
    smooth_pre_kernel<<<BN / 256, 256, 0, stream>>>(ws + OFF_ML, g("gcn.w"), ws + OFF_XW, lab, Nn, BN);
    smooth_deg_kernel<<<BN / 256, 256, 0, stream>>>(er, ev, ws + OFF_EW, lab, ws + OFF_KW, ws + OFF_DIS, BN);
    smooth_gcn_kernel<<<BN / 256, 256, 0, stream>>>(er, ws + OFF_KW, ws + OFF_DIS, ws + OFF_XW,
                                                    g("gcn.b"), out + OUT_LM, sel, BN);

    // 14) hcat = [embed | emb2]  (emb2 computed only for the sampled latent)
    {
        ll total = (ll)Bc * 1088 * Nn;
        copy_ch_kernel<<<(unsigned)((total + 255) / 256), 256, 0, stream>>>(
            out, EB, ws + OFF_BIG, (ll)1152 * Nn, 1088, Nn, total);
    }
    latent_kernel<<<(BN * 64) / 256, 256, 0, stream>>>(out + (ll)576 * Nn, EB,
        g("latent.w"), g("latent.b"), sel, ws + OFF_BIG + (ll)1088 * Nn, (ll)1152 * Nn, Nn, BN);

    // 15) top convs: tc1 -> tc2 -> tc3 (tc3 writes final h output)
    launch_gemm(stream, Bc, 256, 1152, Nn, g("tc1.w"), 1152, 1, 0,
                ws + OFF_BIG, Nn, 1, (ll)1152 * Nn, ws + OFF_P, Nn, 1, (ll)256 * Nn,
                g("tc1.b"), BNPTRS("tb1"), 1);
    launch_gemm(stream, Bc, 256, 256, Nn, g("tc2.w"), 256, 1, 0,
                ws + OFF_P, Nn, 1, (ll)256 * Nn, ws + OFF_Q, Nn, 1, (ll)256 * Nn,
                g("tc2.b"), BNPTRS("tb2"), 1);
    launch_gemm(stream, Bc, 128, 256, Nn, g("tc3.w"), 256, 1, 0,
                ws + OFF_Q, Nn, 1, (ll)256 * Nn, out + OUT_H, Nn, 1, (ll)128 * Nn,
                g("tc3.b"), BNPTRS("tb3"), 1);

    // 16) top classifier; c2 writes (bn,50) directly via output strides
    launch_gemm(stream, Bc, 64, 128, Nn, g("topcls.c1.w"), 128, 1, 0,
                out + OUT_H, Nn, 1, (ll)128 * Nn, ws + OFF_TP1, Nn, 1, (ll)64 * Nn,
                g("topcls.c1.b"), BNPTRS("topcls.b1"), 1);
    launch_gemm(stream, Bc, 50, 64, Nn, g("topcls.c2.w"), 64, 1, 0,
                ws + OFF_TP1, Nn, 1, (ll)64 * Nn, out + OUT_LT, 1, 50, (ll)Nn * 50,
                g("topcls.c2.b"), NOBN, 0);

    #undef BNPTRS
    #undef NOBN
    (void)out_size;
}
